// FasterPointConv_51977694216767
// MI455X (gfx1250) — matrix operand order repaired and checked by reading, not compiled, not run on hardware
//
#include <hip/hip_runtime.h>
#include <hip/hip_bf16.h>
#include <stdint.h>

typedef __attribute__((ext_vector_type(16))) __bf16 v16bf;
typedef __attribute__((ext_vector_type(8)))  __bf16 v8bf;
typedef __attribute__((ext_vector_type(8)))  float  v8f;

#define LRELU(x) ((x) > 0.0f ? (x) : 0.1f * (x))
#define RELU(x)  ((x) > 0.0f ? (x) : 0.0f)

// WMMA wave32 fragment index helpers (CDNA5 ISA 7.12.2, 16-bit operands, 16x16x32)
// A (16x32, lane&15 = M): lanes 0-15 hold K {0..7,16..23}; lanes 16-31 hold K {8..15,24..31}
__device__ __forceinline__ int a_kidx(int lane, int j) {
  return (j & 7) + ((j >> 3) << 4) + ((lane >> 4) << 3);
}
// B (32x16, lane&15 = N): lanes 0-15 hold K {0..7,16..23}; lanes 16-31 hold K {8..15,24..31}
__device__ __forceinline__ int b_kidx(int lane, int j) {
  return (j & 7) + ((j >> 3) << 4) + ((lane >> 4) << 3);
}
// C/D (16x16 f32): row = v + 8*(lane>>4), col = lane&15

// Combine two contiguous 8-element bf16 runs into a v16bf A operand.
__device__ __forceinline__ v16bf combine16(v8bf lo, v8bf hi) {
  return __builtin_shufflevector(lo, hi, 0, 1, 2, 3, 4, 5, 6, 7,
                                 8, 9, 10, 11, 12, 13, 14, 15);
}

// ---------------------------------------------------------------------------
// Kernel 0: fold BN into weights, convert to bf16, and PRE-SWIZZLE into WMMA
// B-fragment order: frag[fragIdx][lane][j], fragIdx = ktile*ctiles + ctile.
// Each lane then loads its whole fragment as one 32-byte vector load.
// ---------------------------------------------------------------------------
struct PrepW {
  const float *u1_w, *u1_b, *u1_s, *u1_h;
  const float *wn_w0, *wn_s0, *wn_w1, *wn_s1, *wn_w2, *wn_s2;
  const float *lin_w, *lin_s;
  const float *u2_w, *u2_b, *u2_s, *u2_h;
  const float *sc_w, *sc_b, *sc_s, *sc_h;
  __bf16 *u1frag;   // [2 kt][2 ct][32][16]  = 2048 bf16
  __bf16 *linfrag;  // [16 kt][4 ct][32][16] = 32768 bf16
  __bf16 *wcfrag;   // [4 kt][8 ct][32][16]  = 16384 bf16 (rows<64:u2, >=64:sc)
  float  *u1bias;   // [32]
  float  *bc;       // [128] combined u2+sc bias
  float  *w0f, *w1f, *w2f; // WeightNet folded fp32
};

__global__ __launch_bounds__(256) void k_prep(PrepW p) {
  const int t = threadIdx.x;
  // u1: K=64, C=32 -> 2 ktiles x 2 ctiles
  for (int e = t; e < 2 * 2 * 512; e += 256) {
    int f = e >> 9, rem = e & 511, lane = rem >> 4, j = rem & 15;
    int kt = f >> 1, ct = f & 1;
    int k = b_kidx(lane, j) + kt * 32, col = ct * 16 + (lane & 15);
    p.u1frag[e] = (__bf16)(p.u1_w[k * 32 + col] * p.u1_s[col]);
  }
  // lin: K=512, C=64 -> 16 ktiles x 4 ctiles
  for (int e = t; e < 16 * 4 * 512; e += 256) {
    int f = e >> 9, rem = e & 511, lane = rem >> 4, j = rem & 15;
    int kt = f >> 2, ct = f & 3;
    int k = b_kidx(lane, j) + kt * 32, col = ct * 16 + (lane & 15);
    p.linfrag[e] = (__bf16)(p.lin_w[k * 64 + col] * p.lin_s[col]);
  }
  // wc = [u2_w*s ; sc_w*s]: K=128, C=128 -> 4 ktiles x 8 ctiles
  for (int e = t; e < 4 * 8 * 512; e += 256) {
    int f = e >> 9, rem = e & 511, lane = rem >> 4, j = rem & 15;
    int kt = f >> 3, ct = f & 7;
    int k = b_kidx(lane, j) + kt * 32, col = ct * 16 + (lane & 15);
    float v = (k < 64) ? p.u2_w[k * 128 + col] * p.u2_s[col]
                       : p.sc_w[(k - 64) * 128 + col] * p.sc_s[col];
    p.wcfrag[e] = (__bf16)v;
  }
  if (t < 32)  p.u1bias[t] = p.u1_b[t] * p.u1_s[t] + p.u1_h[t];
  if (t < 128) p.bc[t] = p.u2_b[t] * p.u2_s[t] + p.u2_h[t] +
                         p.sc_b[t] * p.sc_s[t] + p.sc_h[t];
  if (t < 24)  p.w0f[t] = p.wn_w0[t] * p.wn_s0[t & 7];
  if (t < 64)  p.w1f[t] = p.wn_w1[t] * p.wn_s1[t & 7];
  if (t < 128) p.w2f[t] = p.wn_w2[t] * p.wn_s2[t & 15];
}

// ---------------------------------------------------------------------------
// Kernel 1: unary1  feats_x = leaky(dense_feats @ u1w + b) -> bf16 [N,32]
// also emit dense_feats as bf16 [N,64].  One wave per 16-row tile.
// ---------------------------------------------------------------------------
__global__ __launch_bounds__(32) void k_unary1(const float* __restrict__ feats,
                                               const __bf16* __restrict__ u1frag,
                                               const float* __restrict__ u1bias,
                                               __bf16* __restrict__ feats_x,
                                               __bf16* __restrict__ featsbf) {
  const int tile = blockIdx.x;
  const int lane = threadIdx.x;
  const int r = lane & 15, half = lane >> 4;
  const int row0 = tile * 16;
  __builtin_prefetch(u1frag, 0, 0);

  v8f c0 = {}, c1 = {};
  for (int ks = 0; ks < 2; ++ks) {
    const int klo = (half << 3) + ks * 32;        // contiguous run of 8
    const int khi = 16 + (half << 3) + ks * 32;   // contiguous run of 8
    v8bf alo, ahi;
#pragma unroll
    for (int jj = 0; jj < 8; ++jj) {
      alo[jj] = (__bf16)feats[(size_t)(row0 + r) * 64 + klo + jj];
      ahi[jj] = (__bf16)feats[(size_t)(row0 + r) * 64 + khi + jj];
    }
    // bf16 copy of dense_feats (each (row,k) written exactly once across wave)
    *(v8bf*)&featsbf[(size_t)(row0 + r) * 64 + klo] = alo;
    *(v8bf*)&featsbf[(size_t)(row0 + r) * 64 + khi] = ahi;
    v16bf a  = combine16(alo, ahi);
    v16bf b0 = *(const v16bf*)&u1frag[(ks * 2 + 0) * 512 + lane * 16];
    v16bf b1 = *(const v16bf*)&u1frag[(ks * 2 + 1) * 512 + lane * 16];
    c0 = __builtin_amdgcn_wmma_f32_16x16x32_bf16(false, a, false, b0, (short)0, c0, false, false);
    c1 = __builtin_amdgcn_wmma_f32_16x16x32_bf16(false, a, false, b1, (short)0, c1, false, false);
  }
#pragma unroll
  for (int v = 0; v < 8; ++v) {
    int m = v + (half << 3);
    float x0 = c0[v] + u1bias[r];
    float x1 = c1[v] + u1bias[16 + r];
    feats_x[(size_t)(row0 + m) * 32 + r]      = (__bf16)LRELU(x0);
    feats_x[(size_t)(row0 + m) * 32 + 16 + r] = (__bf16)LRELU(x1);
  }
}

// ---------------------------------------------------------------------------
// Kernel 2: per 16-point tile (4 waves):
//   lane=neighbor: rel-xyz -> output, WeightNet fp32 -> bf16 W^T in LDS
//   einsum G^T(32x32) @ W(32x16) via 2 WMMAs -> 16x512 bf16 LDS tile
//   lin GEMM [16x512]@[512x64] (16 k-steps, 16-col slice per wave) -> ReLU
// ---------------------------------------------------------------------------
struct ConvArgs {
  const float*      xyz;       // [N,3]
  const long long*  nei;       // [N,32] int64
  const __bf16*     feats_x;   // [N,32]
  const float *w0f, *b0, *w1f, *b1, *w2f, *b2;  // WeightNet folded
  const __bf16*     linfrag;   // [16][4][32][16]
  const float*      linb;      // [64] (= lin_h)
  __bf16*           lin_out;   // [N,64]
  float*            loc_out;   // [N,32,3]
};

__global__ __launch_bounds__(128) void k_conv(ConvArgs g) {
  __shared__ __bf16 s_newfeat[16][512];   // 16 KB activation tile
  __shared__ int    s_nei[4][32];
  __shared__ __bf16 s_wgtT[4][16][32];    // transposed: [m][k] -> vector reads

  const int tile = blockIdx.x;
  const int w    = threadIdx.x >> 5;
  const int lane = threadIdx.x & 31;
  const int r    = lane & 15;
  const int half = lane >> 4;
  __builtin_prefetch(g.linfrag, 0, 0);

  for (int it = 0; it < 4; ++it) {
    __syncthreads();                       // protect previous iteration's LDS reads
    const int pi = w * 4 + it;             // point within tile, unique per wave
    const int n  = tile * 16 + pi;

    // ---- neighbor idx + relative xyz + WeightNet (lane == neighbor slot) ----
    const int ni = (int)g.nei[(long long)n * 32 + lane];
    s_nei[w][lane] = ni;
    const float cx = g.xyz[(size_t)n * 3 + 0];
    const float cy = g.xyz[(size_t)n * 3 + 1];
    const float cz = g.xyz[(size_t)n * 3 + 2];
    const float rx = g.xyz[(size_t)ni * 3 + 0] - cx;
    const float ry = g.xyz[(size_t)ni * 3 + 1] - cy;
    const float rz = g.xyz[(size_t)ni * 3 + 2] - cz;
    const long long lb = ((long long)n * 32 + lane) * 3;
    g.loc_out[lb + 0] = rx; g.loc_out[lb + 1] = ry; g.loc_out[lb + 2] = rz;

    float h0[8], h1[8];
#pragma unroll
    for (int j = 0; j < 8; ++j)
      h0[j] = RELU(rx * g.w0f[j] + ry * g.w0f[8 + j] + rz * g.w0f[16 + j] + g.b0[j]);
#pragma unroll
    for (int j = 0; j < 8; ++j) {
      float s = g.b1[j];
#pragma unroll
      for (int i = 0; i < 8; ++i) s += h0[i] * g.w1f[i * 8 + j];
      h1[j] = RELU(s);
    }
#pragma unroll
    for (int j = 0; j < 16; ++j) {
      float s = g.b2[j];
#pragma unroll
      for (int i = 0; i < 8; ++i) s += h1[i] * g.w2f[i * 16 + j];
      s_wgtT[w][j][lane] = (__bf16)RELU(s);   // W^T: [m][k=lane]
    }
    __syncthreads();                       // make s_nei / s_wgtT visible cross-lane

    // ---- einsum: D[c,m] = sum_k feats_x[nei[k]][c] * W[k][m] ----
    // B fragment: lane = m = r, contiguous 16 k values -> one 32B LDS read
    v16bf bfr = *(const v16bf*)&s_wgtT[w][r][half * 16];
    v16bf a0, a1;
#pragma unroll
    for (int j = 0; j < 16; ++j) {
      const int k  = j + (half << 4);
      const int nn = s_nei[w][k];
      a0[j] = g.feats_x[(size_t)nn * 32 + r];
      a1[j] = g.feats_x[(size_t)nn * 32 + 16 + r];
    }
    v8f d0 = {}, d1 = {};
    d0 = __builtin_amdgcn_wmma_f32_16x16x32_bf16(false, a0, false, bfr, (short)0, d0, false, false);
    d1 = __builtin_amdgcn_wmma_f32_16x16x32_bf16(false, a1, false, bfr, (short)0, d1, false, false);
#pragma unroll
    for (int v = 0; v < 8; ++v) {
      const int c = v + (half << 3);
      s_newfeat[pi][c * 16 + r]        = (__bf16)d0[v];   // reshape (c,m) -> c*16+m
      s_newfeat[pi][(c + 16) * 16 + r] = (__bf16)d1[v];
    }
  }
  __syncthreads();

  // ---- lin: [16x512]@[512x64]; wave w owns cols [16w,16w+16) ----
  const int colbase = w << 4;
  v8f acc = {};
  for (int ks = 0; ks < 16; ++ks) {
    v8bf lo = *(const v8bf*)&s_newfeat[r][(half << 3) + ks * 32];
    v8bf hi = *(const v8bf*)&s_newfeat[r][16 + (half << 3) + ks * 32];
    v16bf a = combine16(lo, hi);
    v16bf b = *(const v16bf*)&g.linfrag[(ks * 4 + w) * 512 + lane * 16];
    acc = __builtin_amdgcn_wmma_f32_16x16x32_bf16(false, a, false, b, (short)0, acc, false, false);
  }
#pragma unroll
  for (int v = 0; v < 8; ++v) {
    const int row = tile * 16 + v + (half << 3);
    const int col = colbase + r;
    float x = acc[v] + g.linb[col];
    g.lin_out[(size_t)row * 64 + col] = (__bf16)RELU(x);
  }
}

// ---------------------------------------------------------------------------
// Kernel 3: out = leaky( [lin_out | feats_bf] @ Wc + bc )  -> fp32 [N,128]
// 8 waves per 16-row tile; wave w owns cols [16w,16w+16); K=128 in 4 steps.
// ---------------------------------------------------------------------------
__global__ __launch_bounds__(256) void k_final(const __bf16* __restrict__ lin_out,
                                               const __bf16* __restrict__ featsbf,
                                               const __bf16* __restrict__ wcfrag,
                                               const float* __restrict__ bc,
                                               float* __restrict__ out) {
  const int tile = blockIdx.x;
  const int w    = threadIdx.x >> 5;
  const int lane = threadIdx.x & 31;
  const int r = lane & 15, half = lane >> 4;
  const int colbase = w << 4;
  const int row0 = tile * 16;
  __builtin_prefetch(wcfrag, 0, 0);

  v8f acc = {};
#pragma unroll
  for (int ks = 0; ks < 4; ++ks) {
    const __bf16* __restrict__ src = (ks < 2) ? lin_out : featsbf;
    const int kbase = (ks & 1) * 32 + (half << 3);
    v8bf lo = *(const v8bf*)&src[(size_t)(row0 + r) * 64 + kbase];
    v8bf hi = *(const v8bf*)&src[(size_t)(row0 + r) * 64 + kbase + 16];
    v16bf a = combine16(lo, hi);
    v16bf b = *(const v16bf*)&wcfrag[(ks * 8 + w) * 512 + lane * 16];
    acc = __builtin_amdgcn_wmma_f32_16x16x32_bf16(false, a, false, b, (short)0, acc, false, false);
  }
#pragma unroll
  for (int v = 0; v < 8; ++v) {
    const int row = row0 + v + (half << 3);
    const int col = colbase + r;
    float x = acc[v] + bc[col];
    out[(long long)row * 128 + col] = LRELU(x);
  }
}

// ---------------------------------------------------------------------------
extern "C" void kernel_launch(void* const* d_in, const int* in_sizes, int n_in,
                              void* d_out, int out_size, void* d_ws, size_t ws_size,
                              hipStream_t stream) {
  const float* xyz   = (const float*)d_in[0];
  const float* feats = (const float*)d_in[1];
  const float* u1_w  = (const float*)d_in[2];
  const float* u1_b  = (const float*)d_in[3];
  const float* u1_s  = (const float*)d_in[4];
  const float* u1_h  = (const float*)d_in[5];
  const float* wn_w0 = (const float*)d_in[6];
  const float* wn_s0 = (const float*)d_in[7];
  const float* wn_h0 = (const float*)d_in[8];
  const float* wn_w1 = (const float*)d_in[9];
  const float* wn_s1 = (const float*)d_in[10];
  const float* wn_h1 = (const float*)d_in[11];
  const float* wn_w2 = (const float*)d_in[12];
  const float* wn_s2 = (const float*)d_in[13];
  const float* wn_h2 = (const float*)d_in[14];
  const float* lin_w = (const float*)d_in[15];
  const float* lin_s = (const float*)d_in[16];
  const float* lin_h = (const float*)d_in[17];
  const float* u2_w  = (const float*)d_in[18];
  const float* u2_b  = (const float*)d_in[19];
  const float* u2_s  = (const float*)d_in[20];
  const float* u2_h  = (const float*)d_in[21];
  const float* sc_w  = (const float*)d_in[22];
  const float* sc_b  = (const float*)d_in[23];
  const float* sc_s  = (const float*)d_in[24];
  const float* sc_h  = (const float*)d_in[25];
  const long long* nei = (const long long*)d_in[29];

  const int N = in_sizes[1] / 64;       // 100000; divisible by 16
  const int ntiles = N / 16;

  // workspace carve-out (~32.3 MB)
  char* ws = (char*)d_ws;
  size_t off = 0;
  auto carve = [&](size_t bytes) -> void* {
    void* p = ws + off;
    off = (off + bytes + 255) & ~(size_t)255;
    return p;
  };
  __bf16* feats_x = (__bf16*)carve((size_t)N * 32 * 2);
  __bf16* featsbf = (__bf16*)carve((size_t)N * 64 * 2);
  __bf16* lin_out = (__bf16*)carve((size_t)N * 64 * 2);
  __bf16* u1frag  = (__bf16*)carve(2048 * 2);
  __bf16* linfrag = (__bf16*)carve(32768 * 2);
  __bf16* wcfrag  = (__bf16*)carve(16384 * 2);
  float*  u1bias  = (float*)carve(32 * 4);
  float*  bc      = (float*)carve(128 * 4);
  float*  w0f     = (float*)carve(24 * 4);
  float*  w1f     = (float*)carve(64 * 4);
  float*  w2f     = (float*)carve(128 * 4);

  float* out = (float*)d_out;
  float* loc = out + (size_t)N * 128;   // second output: localized_xyz

  PrepW p = {u1_w, u1_b, u1_s, u1_h,
             wn_w0, wn_s0, wn_w1, wn_s1, wn_w2, wn_s2,
             lin_w, lin_s, u2_w, u2_b, u2_s, u2_h,
             sc_w, sc_b, sc_s, sc_h,
             u1frag, linfrag, wcfrag, u1bias, bc, w0f, w1f, w2f};
  k_prep<<<1, 256, 0, stream>>>(p);

  k_unary1<<<ntiles, 32, 0, stream>>>(feats, u1frag, u1bias, feats_x, featsbf);

  ConvArgs c = {xyz, nei, feats_x,
                w0f, wn_h0, w1f, wn_h1, w2f, wn_h2,
                linfrag, lin_h, lin_out, loc};
  k_conv<<<ntiles, 128, 0, stream>>>(c);

  k_final<<<ntiles, 256, 0, stream>>>(lin_out, featsbf, wcfrag, bc, out);
}
